// Block_46995532153006
// MI455X (gfx1250) — matrix operand order repaired and checked by reading, not compiled, not run on hardware
//
#include <hip/hip_runtime.h>
#include <hip/hip_bf16.h>
#include <math.h>
#include <stdint.h>

// ---------- CDNA5 WMMA types ----------
typedef __attribute__((ext_vector_type(16))) _Float16 v16h;
typedef __attribute__((ext_vector_type(8)))  _Float16 v8h;
typedef __attribute__((ext_vector_type(8)))  float    v8f;

// intra-wave LDS ordering (CDNA5 split counters)
#define LDS_SYNC() asm volatile("s_wait_dscnt 0" ::: "memory")

// ---------- Tensor Data Mover (gfx1250), guarded ----------
#if defined(__HIP_DEVICE_COMPILE__) && __has_builtin(__builtin_amdgcn_tensor_load_to_lds) && __has_builtin(__builtin_amdgcn_s_wait_tensorcnt)
#define USE_TDM 1
#else
#define USE_TDM 0
#endif

#if USE_TDM
typedef __attribute__((ext_vector_type(4))) unsigned int u32x4;
typedef __attribute__((ext_vector_type(8))) int i32x8;
typedef __attribute__((ext_vector_type(4))) int i32x4;

// Load a [tile_rows x tile_cols] f16 tile (row pitch = pitch_elems) from global
// into LDS at lds_off. LDS rows are padded: 16 DWORDs data + 4 DWORDs pad
// (pad_interval code 3, pad_amount code 3) -> 40-half LDS row stride.
__device__ __forceinline__ void tdm_load_tile_f16(const _Float16* gsrc, unsigned lds_off,
                                                  unsigned tile_cols, unsigned tile_rows,
                                                  unsigned long long pitch_elems)
{
  const unsigned long long ga = (unsigned long long)(uintptr_t)gsrc;
  u32x4 g0;
  g0.x = 1u;                                              // count=1 (valid user descriptor)
  g0.y = lds_off;                                         // lds_addr
  g0.z = (unsigned)ga;                                    // global_addr[31:0]
  g0.w = (unsigned)((ga >> 32) & 0x01ffffffull) | (2u << 30);  // global_addr[56:32] | type=2
  i32x8 g1;
  g1[0] = (int)((1u << 16) |        // data_size = 2 bytes
                (1u << 20) |        // pad_enable
                (3u << 22) |        // pad_interval: 16 DWORDs
                (3u << 25));        // pad_amount: 4 DWORDs
  g1[1] = (int)((tile_cols & 0xffffu) << 16);                                  // tensor_dim0[15:0]
  g1[2] = (int)(((tile_cols >> 16) & 0xffffu) | ((tile_rows & 0xffffu) << 16)); // tensor_dim0 hi / dim1 lo
  g1[3] = (int)(((tile_rows >> 16) & 0xffffu) | ((tile_cols & 0xffffu) << 16)); // dim1 hi / tile_dim0
  g1[4] = (int)(tile_rows & 0xffffu);                                          // tile_dim1 ; tile_dim2=0
  g1[5] = (int)(pitch_elems & 0xffffffffull);                                  // tensor_dim0_stride lo
  g1[6] = (int)((pitch_elems >> 32) & 0xffffull);                              // stride hi ; dim1_stride=0
  g1[7] = 0;
  const i32x4 z4 = {0, 0, 0, 0};
  const i32x8 z8 = {0, 0, 0, 0, 0, 0, 0, 0};
  __builtin_amdgcn_tensor_load_to_lds(g0, g1, z4, z4, z8, 0);
}
#endif

__device__ __forceinline__ v16h cat16(v8h lo, v8h hi) {
  return __builtin_shufflevector(lo, hi, 0,1,2,3,4,5,6,7,8,9,10,11,12,13,14,15);
}
__device__ __forceinline__ v8f wmma16(v16h a, v16h b, v8f c) {
  // D = A(16x32 f16) * B(32x16 f16) + C(16x16 f32)
  return __builtin_amdgcn_wmma_f32_16x16x32_f16(false, a, false, b, (short)0, c, false, false);
}

__device__ __forceinline__ float block_reduce256(float v, float* red) {
  const int t = threadIdx.x;
  red[t] = v; __syncthreads();
  for (int o = 128; o > 0; o >>= 1) { if (t < o) red[t] += red[t + o]; __syncthreads(); }
  float r = red[0]; __syncthreads();
  return r;
}

// ---------- precompute the 5 YAT scales ----------
__global__ void scales_kernel(const float* __restrict__ aq, const float* __restrict__ aa,
                              const float* __restrict__ af, const float* __restrict__ ag,
                              const float* __restrict__ ap, float* __restrict__ s)
{
  if (threadIdx.x == 0) {
    s[0] = powf(sqrtf(2304.f) / log1pf(2304.f), aq[0]);
    s[1] = powf(sqrtf(768.f)  / log1pf(768.f),  aa[0]);
    s[2] = powf(sqrtf(3072.f) / log1pf(3072.f), af[0]);
    s[3] = powf(sqrtf(3072.f) / log1pf(3072.f), ag[0]);
    s[4] = powf(sqrtf(768.f)  / log1pf(768.f),  ap[0]);
  }
}

// ---------- LayerNorm -> f16 copy + row norm ----------
__global__ __launch_bounds__(256)
void ln_prep_kernel(const float* __restrict__ x, const float* __restrict__ gamma,
                    _Float16* __restrict__ h16, float* __restrict__ rnorm, int N)
{
  __shared__ float red[256];
  const int row = blockIdx.x, t = threadIdx.x;
  const float* xr = x + (size_t)row * N;
  float s = 0.f;
  for (int i = t; i < N; i += 256) s += xr[i];
  const float mu = block_reduce256(s, red) / (float)N;
  float vv = 0.f;
  for (int i = t; i < N; i += 256) { float d = xr[i] - mu; vv += d * d; }
  const float rstd = rsqrtf(block_reduce256(vv, red) / (float)N + 1e-6f);
  float nn = 0.f;
  _Float16* hr = h16 + (size_t)row * N;
  for (int i = t; i < N; i += 256) {
    float hv = (xr[i] - mu) * rstd * gamma[i];
    hr[i] = (_Float16)hv;
    nn += hv * hv;
  }
  float tot = block_reduce256(nn, red);
  if (t == 0) rnorm[row] = tot;
}

// ---------- fp32 -> f16 copy + row norm ----------
__global__ __launch_bounds__(256)
void prep_kernel(const float* __restrict__ a, _Float16* __restrict__ a16,
                 float* __restrict__ rnorm, int N)
{
  __shared__ float red[256];
  const int row = blockIdx.x, t = threadIdx.x;
  const float* ar = a + (size_t)row * N;
  _Float16* hr = a16 + (size_t)row * N;
  float nn = 0.f;
  for (int i = t; i < N; i += 256) {
    float v = ar[i];
    hr[i] = (_Float16)v;
    nn += v * v;
  }
  float tot = block_reduce256(nn, red);
  if (t == 0) rnorm[row] = tot;
}

// ---------- weight: f16 copy + column norm ----------
__global__ __launch_bounds__(256)
void wprep_kernel(const float* __restrict__ w, _Float16* __restrict__ w16,
                  float* __restrict__ cnorm, int K, int N)
{
  const int n = blockIdx.x * 256 + threadIdx.x;
  if (n >= N) return;
  float s = 0.f;
  for (int k = 0; k < K; ++k) {
    float v = w[(size_t)k * N + n];
    s += v * v;
    w16[(size_t)k * N + n] = (_Float16)v;
  }
  cnorm[n] = s;
}

// ---------- WMMA GEMM with YAT epilogue ----------
// out[m,n] = yat(A@W) (+ optional residual). A:[M,K] f16, W:[K,N] f16.
#define BM 128
#define BN 128
#define BK 32
#define AST 40   // padded LDS strides (40 halves = 80B, multiple of 16B)

__global__ __launch_bounds__(256)
void gemm_yat_kernel(const _Float16* __restrict__ A, const float* __restrict__ an,
                     const _Float16* __restrict__ W, const float* __restrict__ wn,
                     const float* __restrict__ bias, const float* __restrict__ scale_p,
                     const float* __restrict__ resid, float* __restrict__ out,
                     int M, int N, int K)
{
  __shared__ _Float16 As2[2][BM * AST];   // ping-pong A tiles [row][k]
  __shared__ _Float16 Bst[BN * AST];      // B tile, transposed: [n][k]
  const int tid = threadIdx.x, lane = tid & 31, wid = tid >> 5;
  const int wr = wid >> 2, wc = wid & 3;     // 2x4 wave grid
  const int m0 = blockIdx.y * BM, n0 = blockIdx.x * BN;
  const int lh = lane & 15;
  const bool hiHalf = lane >= 16;

  v8f zero = {0.f,0.f,0.f,0.f,0.f,0.f,0.f,0.f};
  v8f acc[4][2];
  #pragma unroll
  for (int mt = 0; mt < 4; ++mt) { acc[mt][0] = zero; acc[mt][1] = zero; }

  const int ntiles = K / BK;

#if USE_TDM
  // prologue: async TDM of first A tile into buffer 0 (one wave issues; EXEC ignored by TDM)
  if (wid == 0) {
    tdm_load_tile_f16(A + (size_t)m0 * K, (unsigned)(uintptr_t)&As2[0][0], BK, BM,
                      (unsigned long long)K);
  }
#endif

  for (int t = 0; t < ntiles; ++t) {
    const int buf = t & 1;
    __syncthreads();   // previous iteration's compute done: LDS reusable

    // stage B tile 32x128, transposed into Bst[n][k] (all threads)
    {
      const int k = tid >> 3, nc = (tid & 7) * 16;
      const _Float16* g = W + (size_t)(t * BK + k) * N + n0 + nc;
      uint4 d0 = *(const uint4*)g, d1 = *(const uint4*)(g + 8);
      const _Float16* h0 = (const _Float16*)&d0;
      const _Float16* h1 = (const _Float16*)&d1;
      #pragma unroll
      for (int j = 0; j < 8; ++j) Bst[(nc + j) * AST + k] = h0[j];
      #pragma unroll
      for (int j = 0; j < 8; ++j) Bst[(nc + 8 + j) * AST + k] = h1[j];
    }

#if USE_TDM
    // prefetch next A tile into the other buffer; wait for the current one
    if (wid == 0) {
      if (t + 1 < ntiles) {
        tdm_load_tile_f16(A + (size_t)m0 * K + (size_t)(t + 1) * BK,
                          (unsigned)(uintptr_t)&As2[buf ^ 1][0], BK, BM,
                          (unsigned long long)K);
        __builtin_amdgcn_s_wait_tensorcnt(1);   // current tile complete, next in flight
      } else {
        __builtin_amdgcn_s_wait_tensorcnt(0);
      }
    }
#else
    // fallback: manual A staging
    {
      const int r = tid >> 1, kc = (tid & 1) * 16;
      const uint4* g = (const uint4*)(A + (size_t)(m0 + r) * K + t * BK + kc);
      uint4 d0 = g[0], d1 = g[1];
      *(uint4*)(&As2[buf][0] + r * AST + kc)     = d0;
      *(uint4*)(&As2[buf][0] + r * AST + kc + 8) = d1;
    }
#endif
    __syncthreads();   // A tile (TDM) + B tile visible to all waves

    const _Float16* As = &As2[buf][0];
    v16h af[4], bf[2];
    const int ka = hiHalf ? 8 : 0;    // A layout: lanes<16 K0-7/K16-23, lanes>=16 K8-15/K24-31
    #pragma unroll
    for (int mt = 0; mt < 4; ++mt) {
      const int r = wr * 64 + mt * 16 + lh;
      v8h a0 = *(const v8h*)(As + r * AST + ka);
      v8h a1 = *(const v8h*)(As + r * AST + ka + 16);
      af[mt] = cat16(a0, a1);
    }
    const int kb = hiHalf ? 8 : 0;    // B layout: lanes<16 K0-7/K16-23, lanes>=16 K8-15/K24-31
    #pragma unroll
    for (int nt = 0; nt < 2; ++nt) {
      const int c = wc * 32 + nt * 16 + lh;
      v8h b0 = *(const v8h*)(Bst + c * AST + kb);
      v8h b1 = *(const v8h*)(Bst + c * AST + kb + 16);
      bf[nt] = cat16(b0, b1);
    }
    #pragma unroll
    for (int mt = 0; mt < 4; ++mt)
      #pragma unroll
      for (int nt = 0; nt < 2; ++nt)
        acc[mt][nt] = wmma16(af[mt], bf[nt], acc[mt][nt]);
  }

  // YAT epilogue: y^2 / (|a|^2 + |w|^2 - 2y + eps) + bias, scaled, (+residual)
  const float scale = scale_p[0];
  #pragma unroll
  for (int mt = 0; mt < 4; ++mt)
    #pragma unroll
    for (int nt = 0; nt < 2; ++nt)
      #pragma unroll
      for (int e = 0; e < 8; ++e) {
        const int gm = m0 + wr * 64 + mt * 16 + e + (hiHalf ? 8 : 0);
        const int gn = n0 + wc * 32 + nt * 16 + lh;
        const float y = acc[mt][nt][e];
        const float dist = an[gm] + wn[gn] - 2.f * y + 1e-6f;
        float r = y * y / dist + bias[gn];
        r *= scale;
        if (resid) r += resid[(size_t)gm * N + gn];
        out[(size_t)gm * N + gn] = r;
      }
}

// ---------- qkv split + elementwise "rope" scaling ----------
__global__ __launch_bounds__(256)
void split_rope_kernel(const float* __restrict__ qkv, _Float16* __restrict__ q,
                       _Float16* __restrict__ k, _Float16* __restrict__ v)
{
  const int idx = blockIdx.x * 256 + threadIdx.x;   // ((b*12+h)*1024+t)*64+d
  if (idx >= 2 * 12 * 1024 * 64) return;
  const int d = idx & 63;
  const int t = (idx >> 6) & 1023;
  const int h = (idx >> 16) % 12;
  const int b = idx / (12 * 1024 * 64);
  const size_t row = (size_t)(b * 1024 + t) * 2304;
  const int col = h * 64 + d;
  const float qv = qkv[row + col];
  const float kv = qkv[row + 768 + col];
  const float vv = qkv[row + 1536 + col];
  const int i2 = (d < 32) ? d : (d - 32);
  const float freq = expf(-9.210340371976184f * (float)(2 * i2) * (1.0f / 64.0f));
  const float ang = (float)t * freq;
  const float f = (d < 32) ? cosf(ang) : sinf(ang);
  q[idx] = (_Float16)(qv * f);
  k[idx] = (_Float16)(kv * f);
  v[idx] = (_Float16)vv;
}

// ---------- flash attention: 1 wave per 16-row query tile ----------
__global__ __launch_bounds__(128)
void attn_kernel(const _Float16* __restrict__ q16, const _Float16* __restrict__ k16,
                 const _Float16* __restrict__ v16, float* __restrict__ o32)
{
  __shared__ float    Sws[4][16][32];
  __shared__ _Float16 Pws[4][16][32];
  __shared__ _Float16 Vst[4][64][40];     // [d][k], stride 40 halves (80B)
  __shared__ float    stats[4][3][16];    // 0:m 1:l 2:c/invl

  const int lane = threadIdx.x & 31;
  const int wid  = threadIdx.x >> 5;
  const int w    = blockIdx.x * 4 + wid;
  const int bh   = w >> 6;                // 0..23
  const int i0   = (w & 63) << 4;
  const int b = bh / 12, h = bh % 12;
  const int lh = lane & 15;
  const bool hiHalf = lane >= 16;
  const float smul = 0.125f;              // 1/sqrt(64)

  v8f zero = {0.f,0.f,0.f,0.f,0.f,0.f,0.f,0.f};
  v8f o[4]; o[0] = o[1] = o[2] = o[3] = zero;

  if (lane < 16) { stats[wid][0][lane] = -1e30f; stats[wid][1][lane] = 0.f; }
  LDS_SYNC();

  const int jtmax = (i0 + 15) >> 5;
  for (int jt = 0; jt <= jtmax; ++jt) {
    const int j0 = jt << 5;

    // stage V[j0..j0+31][0..63] transposed
    {
      const uint4* vp = (const uint4*)(v16 + ((size_t)bh * 1024 + j0 + lane) * 64);
      uint4 tmp[8];
      #pragma unroll
      for (int i = 0; i < 8; ++i) tmp[i] = vp[i];
      const _Float16* th = (const _Float16*)tmp;
      #pragma unroll
      for (int dd = 0; dd < 64; ++dd) Vst[wid][dd][lane] = th[dd];
    }

    // S = Q K^T  (A frags from global q, B frags from global k)
    v8f sacc[2]; sacc[0] = zero; sacc[1] = zero;
    #pragma unroll
    for (int sh = 0; sh < 2; ++sh) {
      const size_t qb = ((size_t)bh * 1024 + i0 + lh) * 64 + sh * 32 + (hiHalf ? 8 : 0);
      v8h a0 = *(const v8h*)(q16 + qb);
      v8h a1 = *(const v8h*)(q16 + qb + 16);
      v16h af = cat16(a0, a1);
      #pragma unroll
      for (int nt = 0; nt < 2; ++nt) {
        const size_t kb = ((size_t)bh * 1024 + j0 + nt * 16 + lh) * 64 + sh * 32 + (hiHalf ? 8 : 0);
        v8h b0 = *(const v8h*)(k16 + kb);
        v8h b1 = *(const v8h*)(k16 + kb + 16);
        sacc[nt] = wmma16(af, cat16(b0, b1), sacc[nt]);
      }
    }

    // write masked, scaled scores
    #pragma unroll
    for (int nt = 0; nt < 2; ++nt)
      #pragma unroll
      for (int e = 0; e < 8; ++e) {
        const int rr = e + (hiHalf ? 8 : 0);
        const int cc = nt * 16 + lh;
        Sws[wid][rr][cc] = ((j0 + cc) <= (i0 + rr)) ? sacc[nt][e] * smul : -1e30f;
      }
    LDS_SYNC();

    // online softmax: one lane per row
    if (lane < 16) {
      const int r = lane;
      float mo = stats[wid][0][r];
      float mt = -1e30f;
      #pragma unroll
      for (int j = 0; j < 32; ++j) mt = fmaxf(mt, Sws[wid][r][j]);
      const float mn = fmaxf(mo, mt);
      const float c = expf(mo - mn);
      float l = stats[wid][1][r] * c;
      #pragma unroll
      for (int j = 0; j < 32; ++j) {
        const float p = expf(Sws[wid][r][j] - mn);
        l += p;
        Pws[wid][r][j] = (_Float16)p;
      }
      stats[wid][0][r] = mn;
      stats[wid][1][r] = l;
      stats[wid][2][r] = c;
    }
    LDS_SYNC();

    // rescale running output
    float cf[8];
    #pragma unroll
    for (int e = 0; e < 8; ++e) cf[e] = stats[wid][2][e + (hiHalf ? 8 : 0)];
    #pragma unroll
    for (int nt = 0; nt < 4; ++nt)
      #pragma unroll
      for (int e = 0; e < 8; ++e) o[nt][e] *= cf[e];

    // O += P @ V
    {
      const int kh = hiHalf ? 8 : 0;
      v8h a0 = *(const v8h*)(&Pws[wid][lh][kh]);
      v8h a1 = *(const v8h*)(&Pws[wid][lh][kh + 16]);
      v16h af = cat16(a0, a1);
      const int kv = hiHalf ? 8 : 0;
      #pragma unroll
      for (int nt = 0; nt < 4; ++nt) {
        const int dcol = nt * 16 + lh;
        v8h b0 = *(const v8h*)(&Vst[wid][dcol][kv]);
        v8h b1 = *(const v8h*)(&Vst[wid][dcol][kv + 16]);
        o[nt] = wmma16(af, cat16(b0, b1), o[nt]);
      }
    }
    LDS_SYNC();
  }

  if (lane < 16) stats[wid][2][lane] = 1.f / stats[wid][1][lane];
  LDS_SYNC();
  float inv[8];
  #pragma unroll
  for (int e = 0; e < 8; ++e) inv[e] = stats[wid][2][e + (hiHalf ? 8 : 0)];
  #pragma unroll
  for (int nt = 0; nt < 4; ++nt)
    #pragma unroll
    for (int e = 0; e < 8; ++e) {
      const int rr = e + (hiHalf ? 8 : 0);
      const int dd = nt * 16 + lh;
      o32[((size_t)b * 1024 + i0 + rr) * 768 + h * 64 + dd] = o[nt][e] * inv[e];
    }
}

// ---------- u = gelu(gate_out) * fc_out -> f16 + row norm ----------
__global__ __launch_bounds__(256)
void gelu_gate_kernel(const float* __restrict__ ug, const float* __restrict__ gg,
                      _Float16* __restrict__ u16, float* __restrict__ rnorm, int N)
{
  __shared__ float red[256];
  const int row = blockIdx.x, t = threadIdx.x;
  const float* ar = ug + (size_t)row * N;
  const float* br = gg + (size_t)row * N;
  _Float16* ur = u16 + (size_t)row * N;
  float nn = 0.f;
  for (int i = t; i < N; i += 256) {
    const float a = ar[i];
    const float ge = 0.5f * a * (1.f + tanhf(0.7978845608028654f * (a + 0.044715f * a * a * a)));
    const float u = ge * br[i];
    ur[i] = (_Float16)u;
    nn += u * u;
  }
  float tot = block_reduce256(nn, red);
  if (t == 0) rnorm[row] = tot;
}

// ---------- host orchestration ----------
extern "C" void kernel_launch(void* const* d_in, const int* in_sizes, int n_in,
                              void* d_out, int out_size, void* d_ws, size_t ws_size,
                              hipStream_t stream)
{
  (void)in_sizes; (void)n_in; (void)out_size; (void)ws_size;
  const float* x    = (const float*)d_in[0];
  // d_in[1] = mask (causal, computed analytically)
  const float* ln1  = (const float*)d_in[2];
  const float* wqkv = (const float*)d_in[3];
  const float* bqkv = (const float*)d_in[4];
  const float* aqkv = (const float*)d_in[5];
  const float* wao  = (const float*)d_in[6];
  const float* bao  = (const float*)d_in[7];
  const float* aao  = (const float*)d_in[8];
  const float* ln2  = (const float*)d_in[9];
  const float* wfc  = (const float*)d_in[10];
  const float* bfc  = (const float*)d_in[11];
  const float* afc  = (const float*)d_in[12];
  const float* wg   = (const float*)d_in[13];
  const float* bg   = (const float*)d_in[14];
  const float* ag   = (const float*)d_in[15];
  const float* wp   = (const float*)d_in[16];
  const float* bp   = (const float*)d_in[17];
  const float* ap   = (const float*)d_in[18];

  const int M = 2048, C = 768, NQ = 2304, HID = 3072;
  const int QKVE = 2 * 12 * 1024 * 64;   // per-tensor q/k/v elements

  char* cur = (char*)d_ws;
  auto alloc = [&](size_t bytes) -> void* {
    void* p = (void*)cur;
    cur += (bytes + 255) & ~(size_t)255;
    return p;
  };

  float*    scl   = (float*)   alloc(8 * 4);
  _Float16* h16   = (_Float16*)alloc((size_t)M * C * 2);
  float*    hn    = (float*)   alloc((size_t)M * 4);
  _Float16* wq16  = (_Float16*)alloc((size_t)C * NQ * 2);
  float*    wqn   = (float*)   alloc((size_t)NQ * 4);
  float*    qkv32 = (float*)   alloc((size_t)M * NQ * 4);
  _Float16* q16   = (_Float16*)alloc((size_t)QKVE * 2);
  _Float16* k16   = (_Float16*)alloc((size_t)QKVE * 2);
  _Float16* v16b  = (_Float16*)alloc((size_t)QKVE * 2);
  float*    o32   = (float*)   alloc((size_t)M * C * 4);
  _Float16* o16   = (_Float16*)alloc((size_t)M * C * 2);
  float*    onrm  = (float*)   alloc((size_t)M * 4);
  _Float16* wao16 = (_Float16*)alloc((size_t)C * C * 2);
  float*    waon  = (float*)   alloc((size_t)C * 4);
  float*    x1    = (float*)   alloc((size_t)M * C * 4);
  _Float16* h2    = (_Float16*)alloc((size_t)M * C * 2);
  float*    h2n   = (float*)   alloc((size_t)M * 4);
  _Float16* wfc16 = (_Float16*)alloc((size_t)C * HID * 2);
  float*    wfcn  = (float*)   alloc((size_t)HID * 4);
  _Float16* wg16  = (_Float16*)alloc((size_t)C * HID * 2);
  float*    wgn   = (float*)   alloc((size_t)HID * 4);
  float*    fc32  = (float*)   alloc((size_t)M * HID * 4);
  float*    g32   = (float*)   alloc((size_t)M * HID * 4);
  _Float16* u16   = (_Float16*)alloc((size_t)M * HID * 2);
  float*    un    = (float*)   alloc((size_t)M * 4);
  _Float16* wp16  = (_Float16*)alloc((size_t)HID * C * 2);
  float*    wpn   = (float*)   alloc((size_t)C * 4);

  // scales + weight prep (f16 + column norms)
  scales_kernel<<<1, 32, 0, stream>>>(aqkv, aao, afc, ag, ap, scl);
  wprep_kernel<<<NQ / 256, 256, 0, stream>>>(wqkv, wq16, wqn, C, NQ);
  wprep_kernel<<<C / 256, 256, 0, stream>>>(wao, wao16, waon, C, C);
  wprep_kernel<<<HID / 256, 256, 0, stream>>>(wfc, wfc16, wfcn, C, HID);
  wprep_kernel<<<HID / 256, 256, 0, stream>>>(wg, wg16, wgn, C, HID);
  wprep_kernel<<<C / 256, 256, 0, stream>>>(wp, wp16, wpn, HID, C);

  // LN1 -> h ; qkv = yat(h @ w_qkv)
  ln_prep_kernel<<<M, 256, 0, stream>>>(x, ln1, h16, hn, C);
  gemm_yat_kernel<<<dim3(NQ / 128, M / 128), 256, 0, stream>>>(
      h16, hn, wq16, wqn, bqkv, scl + 0, nullptr, qkv32, M, NQ, C);

  // split + rope, attention
  split_rope_kernel<<<QKVE / 256, 256, 0, stream>>>(qkv32, q16, k16, v16b);
  attn_kernel<<<(2 * 12 * 64) / 4, 128, 0, stream>>>(q16, k16, v16b, o32);

  // x1 = x + yat(o @ w_ao)
  prep_kernel<<<M, 256, 0, stream>>>(o32, o16, onrm, C);
  gemm_yat_kernel<<<dim3(C / 128, M / 128), 256, 0, stream>>>(
      o16, onrm, wao16, waon, bao, scl + 1, x, x1, M, C, C);

  // LN2 -> h2 ; gate = yat(h2@w_fc) ; u0 = yat(h2@w_gate)
  ln_prep_kernel<<<M, 256, 0, stream>>>(x1, ln2, h2, h2n, C);
  gemm_yat_kernel<<<dim3(HID / 128, M / 128), 256, 0, stream>>>(
      h2, h2n, wfc16, wfcn, bfc, scl + 2, nullptr, fc32, M, HID, C);
  gemm_yat_kernel<<<dim3(HID / 128, M / 128), 256, 0, stream>>>(
      h2, h2n, wg16, wgn, bg, scl + 3, nullptr, g32, M, HID, C);

  // u = gelu(u0) * gate ; out = x1 + yat(u @ w_proj)
  gelu_gate_kernel<<<M, 256, 0, stream>>>(g32, fc32, u16, un, HID);
  gemm_yat_kernel<<<dim3(C / 128, M / 128), 256, 0, stream>>>(
      u16, un, wp16, wpn, bp, scl + 4, x1, (float*)d_out, M, C, HID);
}
